// GIBLi_KPFCNN_35648228556896
// MI455X (gfx1250) — matrix operand order repaired and checked by reading, not compiled, not run on hardware
//
#include <hip/hip_runtime.h>

// ===========================================================================
// KPFCNN forward for MI455X (gfx1250, wave32, WMMA).
// - Dense contractions -> v_wmma_f32_16x16x32_bf16 (bf16 operands, f32 accum).
// - GEMM A-tiles staged with CDNA5 async copy (global_load_async_to_lds_b128,
//   ASYNCcnt-tracked), B-tiles transposed into LDS with vector global reads.
// - All GEMM K dims padded to %32==0 (only enc1_1's 75 needed padding -> 96,
//   final head's 10 cols -> 16), so tile staging is unguarded 16B transfers.
// - Gathers / GroupNorm / residuals stay f32: the network moves ~1.5 GB of
//   irregular gather traffic vs ~40 GFLOP of GEMM -> HBM-bound at 23.3 TB/s,
//   bf16 operands halve GEMM bytes at no practical accuracy cost.
// ===========================================================================

typedef __attribute__((ext_vector_type(16))) __bf16 v16bf;
typedef __attribute__((ext_vector_type(8)))  float  v8f;

#define MKP 15   // kernel points

__device__ __forceinline__ unsigned short f2bf(float f) {
  unsigned u = __float_as_uint(f);
  unsigned r = 0x7FFFu + ((u >> 16) & 1u);   // round-to-nearest-even
  return (unsigned short)((u + r) >> 16);
}

// ---------------------------------------------------------------------------
// f32 -> bf16 cast (grid-stride)
// ---------------------------------------------------------------------------
__global__ void k_cast_bf16(const float* __restrict__ src,
                            unsigned short* __restrict__ dst, long n) {
  long i = (long)blockIdx.x * blockDim.x + threadIdx.x;
  long st = (long)gridDim.x * blockDim.x;
  for (; i < n; i += st) dst[i] = f2bf(src[i]);
}

// f32 [srcR, srcC] -> bf16 [dstR, dstC], zero-padded (for K/N padding)
__global__ void k_cast_pad2d(const float* __restrict__ src,
                             unsigned short* __restrict__ dst,
                             int srcR, int srcC, int dstR, int dstC) {
  long n = (long)dstR * dstC;
  long i = (long)blockIdx.x * blockDim.x + threadIdx.x;
  long st = (long)gridDim.x * blockDim.x;
  for (; i < n; i += st) {
    int r = (int)(i / dstC), c = (int)(i - (long)r * dstC);
    dst[i] = (r < srcR && c < srcC) ? f2bf(src[(long)r * srcC + c])
                                    : (unsigned short)0;
  }
}

// ---------------------------------------------------------------------------
// GEMM: C[Nr,NcOut] = A[Nr,K] * B[K,NcPad] (+bias), bf16 in / f32 out.
// Preconditions: Nr%64==0, K%32==0, NcPad%8==0, A/B 16B-aligned.
// 64x64 tile per 128-thread block (4 waves); wave owns 16 rows x four 16x16
// WMMA accumulators.  A tile: async global->LDS b128 copies (ASYNCcnt).
// B tile: uint4 global reads, transposed into LDS ([n][k]) so both operand
// fragments are contiguous 16B chunks (ds_load_b128).
// Fragment layouts per CDNA5 ISA 7.12.2 (16-bit A 16x32 / B 32x16).
// ---------------------------------------------------------------------------
#define GTM 64
#define GTN 64
#define GTK 32
#define GLS 40   // LDS row stride in halves (80B: keeps 16B chunk alignment)

__global__ void __launch_bounds__(128)
k_gemm_bf16(const unsigned short* __restrict__ A,
            const unsigned short* __restrict__ B,
            float* __restrict__ C, const float* __restrict__ bias,
            int Nr, int K, int NcPad, int NcOut) {
  __shared__ __align__(16) unsigned short Alds[GTM * GLS];  // [row][k]
  __shared__ __align__(16) unsigned short Blds[GTN * GLS];  // [n][k] (transposed)

  const int tid  = threadIdx.x;
  const int lane = tid & 31;
  const int wave = tid >> 5;
  const int rowB = blockIdx.y * GTM;
  const int colB = blockIdx.x * GTN;
  const int m    = lane & 15;
  const int hs   = lane >> 4;          // half-wave select
  const int kb   = hs << 4;

  v8f acc[4];
  for (int s = 0; s < 4; ++s)
    for (int j = 0; j < 8; ++j) acc[s][j] = 0.0f;

  union Frag { uint4 q[2]; v16bf v; };

  for (int k0 = 0; k0 < K; k0 += GTK) {
    // ---- A tile: 64x32 halves, 256 x 16B chunks via async copy to LDS ----
#pragma unroll
    for (int it = 0; it < 2; ++it) {
      int c = tid + it * 128;
      int r = c >> 2, q = c & 3;                 // row, 8-half chunk
      unsigned lds = (unsigned)(size_t)(&Alds[r * GLS + q * 8]);
      const unsigned short* gp = A + ((long)(rowB + r) * K + k0 + q * 8);
      asm volatile("global_load_async_to_lds_b128 %0, %1, off"
                   :: "v"(lds), "v"(gp) : "memory");
    }
    // ---- B tile: 32x64 halves, vector reads + transpose store ----
#pragma unroll
    for (int it = 0; it < 2; ++it) {
      int c = tid + it * 128;
      int kk = c >> 3, nc = (c & 7) * 8;         // k row, 8-col chunk
      int gc = colB + nc;
      if (gc + 8 <= NcPad) {
        uint4 v4 = *reinterpret_cast<const uint4*>(
            B + ((long)(k0 + kk) * NcPad + gc));
        const unsigned short* h = reinterpret_cast<const unsigned short*>(&v4);
#pragma unroll
        for (int j = 0; j < 8; ++j) Blds[(nc + j) * GLS + kk] = h[j];
      } else if (k0 == 0) {                      // zero once; stays zero
#pragma unroll
        for (int j = 0; j < 8; ++j) Blds[(nc + j) * GLS + kk] = 0;
      }
    }
    asm volatile("s_wait_asynccnt 0x0" ::: "memory");
    __syncthreads();

    // ---- operand fragments (all loads first, then WMMAs back-to-back) ----
    // A frag: lane row m; VGPR v holds k = (v/4)*16 + hs*8 + (v%4)*2 (+0,+1)
    //   => halves [hs*8, hs*8+8) and [16+hs*8, 16+hs*8+8): two 16B chunks.
    Frag fa;
    const int ar = wave * 16 + m;
    fa.q[0] = *reinterpret_cast<const uint4*>(&Alds[ar * GLS + hs * 8]);
    fa.q[1] = *reinterpret_cast<const uint4*>(&Alds[ar * GLS + 16 + hs * 8]);
    // B frag: lane col n; VGPR v holds k = (v/4)*16 + hs*8 + (v%4)*2 (+0,+1)
    //   => halves [hs*8, hs*8+8) and [16+hs*8, 16+hs*8+8) in transposed LDS row.
    Frag fb[4];
#pragma unroll
    for (int s = 0; s < 4; ++s) {
      const int bn = s * 16 + m;
      fb[s].q[0] = *reinterpret_cast<const uint4*>(&Blds[bn * GLS + hs * 8]);
      fb[s].q[1] = *reinterpret_cast<const uint4*>(&Blds[bn * GLS + 16 + hs * 8]);
    }
#pragma unroll
    for (int s = 0; s < 4; ++s)
      acc[s] = __builtin_amdgcn_wmma_f32_16x16x32_bf16(
          false, fa.v, false, fb[s].v, (short)0, acc[s], false, false);
    __syncthreads();
  }

  // C/D layout: VGPR j -> M = j + 8*hs, N = lane%16
#pragma unroll
  for (int s = 0; s < 4; ++s) {
    int gc = colB + s * 16 + m;
    if (gc < NcOut) {
      float bv = bias ? bias[gc] : 0.0f;
#pragma unroll
      for (int j = 0; j < 8; ++j) {
        int gr = rowB + wave * 16 + j + (hs << 3);
        C[(long)gr * NcOut + gc] = acc[s][j] + bv;
      }
    }
  }
}

// ---------------------------------------------------------------------------
// KPConv gather: wf[n, m, c] = sum_k max(0, 1 - |rel_k - kp_m|/sigma) * feat[nb,c]
// One block per query point: neighbor rel-positions + influence table in LDS,
// channels across lanes (coalesced feature gathers), 15 f32 accumulators.
// Output bf16 rows of `rowStride` halves (>= MKP*C, %32==0), tail zeroed.
// ---------------------------------------------------------------------------
__global__ void k_kpconv_wf(const float* __restrict__ qpts,
                            const float* __restrict__ spts,
                            const int* __restrict__ nb,
                            const float* __restrict__ kpts,
                            const float* __restrict__ sfeats,
                            unsigned short* __restrict__ wf,
                            int K, int C, int rowStride, float inv_sigma) {
  extern __shared__ float smem[];
  float* rel  = smem;               // [K][3]
  float* infl = smem + K * 3;       // [K][MKP]
  int*   nbs  = (int*)(infl + K * MKP);

  const int n = blockIdx.x;
  const int tid = threadIdx.x, bs = blockDim.x;
  const float qx = qpts[n * 3], qy = qpts[n * 3 + 1], qz = qpts[n * 3 + 2];

  for (int k = tid; k < K; k += bs) {
    int j = nb[(long)n * K + k];
    nbs[k] = j;
    rel[k * 3 + 0] = spts[j * 3 + 0] - qx;
    rel[k * 3 + 1] = spts[j * 3 + 1] - qy;
    rel[k * 3 + 2] = spts[j * 3 + 2] - qz;
  }
  __syncthreads();
  for (int i = tid; i < K * MKP; i += bs) {
    int k = i / MKP, mm = i - k * MKP;
    float dx = rel[k * 3 + 0] - kpts[mm * 3 + 0];
    float dy = rel[k * 3 + 1] - kpts[mm * 3 + 1];
    float dz = rel[k * 3 + 2] - kpts[mm * 3 + 2];
    float d = sqrtf(dx * dx + dy * dy + dz * dz);
    infl[i] = fmaxf(0.0f, 1.0f - d * inv_sigma);
  }
  __syncthreads();
  unsigned short* orow = wf + (long)n * rowStride;
  for (int c = tid; c < C; c += bs) {
    float acc[MKP];
#pragma unroll
    for (int mm = 0; mm < MKP; ++mm) acc[mm] = 0.0f;
    for (int k = 0; k < K; ++k) {
      float f = sfeats[(long)nbs[k] * C + c];
      const float* ik = &infl[k * MKP];
#pragma unroll
      for (int mm = 0; mm < MKP; ++mm) acc[mm] = fmaf(ik[mm], f, acc[mm]);
    }
#pragma unroll
    for (int mm = 0; mm < MKP; ++mm) orow[(long)mm * C + c] = f2bf(acc[mm]);
  }
  for (int z = MKP * C + tid; z < rowStride; z += bs) orow[z] = 0;  // K pad
}

// ---------------------------------------------------------------------------
// GroupNorm (8 groups) + optional pre-bias + activation (0 none, 1 leaky, 2 relu)
// One row per block; one wave32 per group; shfl_xor reductions.
// ---------------------------------------------------------------------------
__global__ void __launch_bounds__(256)
k_gn_act(const float* __restrict__ x, float* __restrict__ y,
         const float* __restrict__ gamma, const float* __restrict__ beta,
         const float* __restrict__ prebias, int C, int act) {
  const int row  = blockIdx.x;
  const int lane = threadIdx.x & 31;
  const int g    = threadIdx.x >> 5;      // 8 waves == 8 groups
  const int gsz  = C >> 3;
  const long base = (long)row * C + (long)g * gsz;
  const int cb = g * gsz;

  float s = 0.0f, ss = 0.0f;
  for (int i = lane; i < gsz; i += 32) {
    float v = x[base + i];
    if (prebias) v += prebias[cb + i];
    s += v; ss += v * v;
  }
  for (int o = 16; o >= 1; o >>= 1) {
    s  += __shfl_xor(s, o, 32);
    ss += __shfl_xor(ss, o, 32);
  }
  float mean = s / (float)gsz;
  float var  = ss / (float)gsz - mean * mean;
  float rstd = rsqrtf(var + 1e-5f);
  for (int i = lane; i < gsz; i += 32) {
    float v = x[base + i];
    if (prebias) v += prebias[cb + i];
    v = (v - mean) * rstd * gamma[cb + i] + beta[cb + i];
    if (act == 1) v = (v > 0.0f) ? v : 0.1f * v;
    else if (act == 2) v = fmaxf(v, 0.0f);
    y[base + i] = v;
  }
}

// out[n,c] = max_k src[nb[n,k], c]   (strided shortcut max-pool)
__global__ void k_rowmax(const float* __restrict__ src, const int* __restrict__ nb,
                         float* __restrict__ out, int K, int C) {
  __shared__ int nbs[64];
  const int n = blockIdx.x;
  for (int k = threadIdx.x; k < K; k += blockDim.x) nbs[k] = nb[(long)n * K + k];
  __syncthreads();
  for (int c = threadIdx.x; c < C; c += blockDim.x) {
    float mv = -3.4e38f;
    for (int k = 0; k < K; ++k) mv = fmaxf(mv, src[(long)nbs[k] * C + c]);
    out[(long)n * C + c] = mv;
  }
}

// out = leaky(a + b)
__global__ void k_add_leaky(const float* __restrict__ a, const float* __restrict__ b,
                            float* __restrict__ o, long n) {
  long i = (long)blockIdx.x * blockDim.x + threadIdx.x;
  long st = (long)gridDim.x * blockDim.x;
  for (; i < n; i += st) {
    float v = a[i] + b[i];
    o[i] = (v > 0.0f) ? v : 0.1f * v;
  }
}

// dst[n] = bf16( lat[up[n]] || skip[n] )   (decoder upsample + concat)
__global__ void k_concat_gather(const float* __restrict__ lat, const int* __restrict__ up,
                                const float* __restrict__ skip,
                                unsigned short* __restrict__ dst, int c1, int c2) {
  const int n = blockIdx.x;
  const int s = up[n];
  const long ob = (long)n * (c1 + c2);
  for (int c = threadIdx.x; c < c1 + c2; c += blockDim.x) {
    float v = (c < c1) ? lat[(long)s * c1 + c] : skip[(long)n * c2 + (c - c1)];
    dst[ob + c] = f2bf(v);
  }
}

// ===========================================================================
// Host orchestration
// ===========================================================================
extern "C" void kernel_launch(void* const* d_in, const int* in_sizes, int n_in,
                              void* d_out, int out_size, void* d_ws, size_t ws_size,
                              hipStream_t stream) {
  (void)in_sizes; (void)n_in; (void)out_size; (void)ws_size;

  static const int NS[5]   = {65536, 16384, 4096, 1024, 256};
  static const int KSA[5]  = {24, 40, 34, 35, 34};
  const float SIG1 = 0.08f;

  // --- input index map -----------------------------------------------------
  // Assumes JAX pytree flattening (dict keys sorted) of setup_inputs():
  //   consts(14) | feats | neighbors(5) | params(153) | points(5) | sub(4) | up(4)
  // params sorted: cls(6: b,b1,b2,g,w1,w2) dec1..4(3 each: b,g,w)
  //   enc1_1(3: b,g,kp_w) then 13 blocks (kp_b,kp_g,kp_w,[sc_b,sc_g,sc_w],
  //   u1_b,u1_g,u1_w,u2_b,u2_g,u2_w)
  const int IDX_CONST = 0;                      // enc1_1, then blocks in order
  const int IDX_FEATS = 14;
  const int IDX_NBR   = 15;
  const int IDX_CLS_B = 20, IDX_CLS_B1 = 21, IDX_CLS_B2 = 22;
  const int IDX_CLS_G = 23, IDX_CLS_W1 = 24, IDX_CLS_W2 = 25;
  const int IDX_DEC1  = 26;                     // dec1 b,g,w ; dec2 +3 ; dec3 +6 ; dec4 +9
  const int IDX_E11_B = 38, IDX_E11_G = 39, IDX_E11_KPW = 40;
  const int IDX_PTS   = 173;
  const int IDX_SUB   = 178;
  const int IDX_UP    = 182;

  struct Blk { int qs, ss, cin, cout, strided, has_sc; float mult; int pb; };
  const Blk blk[13] = {
    {0,0,  64, 128,0,1, 1.f, 41}, {1,0, 128, 128,1,0, 1.f, 53},
    {1,1, 128, 256,0,1, 2.f, 62}, {1,1, 256, 256,0,0, 2.f, 74},
    {2,1, 256, 256,1,0, 2.f, 83}, {2,2, 256, 512,0,1, 4.f, 92},
    {2,2, 512, 512,0,0, 4.f,104}, {3,2, 512, 512,1,0, 4.f,113},
    {3,3, 512,1024,0,1, 8.f,122}, {3,3,1024,1024,0,0, 8.f,134},
    {4,3,1024,1024,1,0, 8.f,143}, {4,4,1024,2048,0,1,16.f,152},
    {4,4,2048,2048,0,0,16.f,164}};
  const int stageFinal[13] = {1,0,0,1,0,0,1,0,0,1,0,0,1};

  auto F = [&](int i) { return (const float*)d_in[i]; };
  auto I = [&](int i) { return (const int*)d_in[i]; };

  // --- workspace bump allocator -------------------------------------------
  char* wsb = (char*)d_ws;
  size_t off = 0;
  auto alloc = [&](size_t bytes) -> void* {
    off = (off + 255) & ~(size_t)255;
    void* p = wsb + off;
    off += bytes;
    return p;
  };
  auto gsz = [](long n) {
    long b = (n + 255) / 256;
    return (int)(b > 4096 ? 4096 : (b < 1 ? 1 : b));
  };

  // --- kernel launch helpers ----------------------------------------------
  auto castf = [&](const float* src, unsigned short* dst, long n) {
    k_cast_bf16<<<gsz(n), 256, 0, stream>>>(src, dst, n);
  };
  auto castw = [&](int idx, long n) -> const unsigned short* {
    unsigned short* p = (unsigned short*)alloc(n * 2);
    castf(F(idx), p, n);
    return p;
  };
  auto castw_pad = [&](int idx, int sr, int sc, int dr, int dc)
      -> const unsigned short* {
    unsigned short* p = (unsigned short*)alloc((long)dr * dc * 2);
    k_cast_pad2d<<<gsz((long)dr * dc), 256, 0, stream>>>(F(idx), p, sr, sc, dr, dc);
    return p;
  };
  auto gemm = [&](const unsigned short* A, const unsigned short* B, float* C,
                  int Nr, int K, int NcPad, int NcOut, const float* bias) {
    dim3 g((NcPad + GTN - 1) / GTN, (Nr + GTM - 1) / GTM);
    k_gemm_bf16<<<g, 128, 0, stream>>>(A, B, C, bias, Nr, K, NcPad, NcOut);
  };
  auto gn = [&](const float* x, float* y, const float* gm, const float* bt,
                const float* pb, int N, int C, int act) {
    k_gn_act<<<N, 256, 0, stream>>>(x, y, gm, bt, pb, C, act);
  };
  auto kpconv = [&](const float* qp, const float* sp, const int* nb,
                    const float* kp, const float* sf, unsigned short* wf,
                    int Nq, int K, int C, int rowStride, float sigma) {
    int bs = (C <= 64) ? 64 : (C <= 128 ? 128 : 256);
    size_t sm = (size_t)K * (3 + MKP) * 4 + (size_t)K * 4;
    k_kpconv_wf<<<Nq, bs, sm, stream>>>(qp, sp, nb, kp, sf, wf, K, C,
                                        rowStride, 1.0f / sigma);
  };

  // --- pre-cast all GEMM weights to bf16 (K-pad enc1_1, N-pad cls w2) -----
  const unsigned short* w_e11 = castw_pad(IDX_E11_KPW, MKP * 5, 64, 96, 64);
  const unsigned short *u1w[13], *kpw[13], *u2w[13], *scw[13];
  int o_u1g[13], o_u1b[13], o_kpg[13], o_kpb[13], o_u2g[13], o_u2b[13];
  int o_scg[13], o_scb[13];
  for (int b = 0; b < 13; ++b) {
    const Blk& B = blk[b];
    const int cm = B.cout / 4;
    int pb = B.pb;
    o_kpb[b] = pb + 0; o_kpg[b] = pb + 1;
    kpw[b] = castw(pb + 2, (long)MKP * cm * cm);
    int u = B.has_sc ? pb + 6 : pb + 3;
    if (B.has_sc) {
      o_scb[b] = pb + 3; o_scg[b] = pb + 4;
      scw[b] = castw(pb + 5, (long)B.cin * B.cout);
    } else { o_scb[b] = o_scg[b] = 0; scw[b] = nullptr; }
    o_u1b[b] = u + 0; o_u1g[b] = u + 1;
    u1w[b] = castw(u + 2, (long)B.cin * cm);
    o_u2b[b] = u + 3; o_u2g[b] = u + 4;
    u2w[b] = castw(u + 5, (long)cm * B.cout);
  }
  const long dec_cin[4]  = {384, 768, 1536, 3072};
  const long dec_cout[4] = {128, 256, 512, 1024};
  const unsigned short* decw[4];
  for (int i = 0; i < 4; ++i)
    decw[i] = castw(IDX_DEC1 + 3 * i + 2, dec_cin[i] * dec_cout[i]);
  const unsigned short* w_cls1 = castw(IDX_CLS_W1, 128L * 64);
  const unsigned short* w_cls2 = castw_pad(IDX_CLS_W2, 64, 10, 64, 16);

  // --- activation / scratch buffers ---------------------------------------
  unsigned short* actbf = (unsigned short*)alloc(27L * 1024 * 1024 * 2);
  unsigned short* wfbf  = (unsigned short*)alloc((long)65536 * MKP * 32 * 2);
  float* t0  = (float*)alloc(9L * 1024 * 1024 * 4);
  float* t1  = (float*)alloc(4L * 1024 * 1024 * 4);
  float* t2  = (float*)alloc(9L * 1024 * 1024 * 4);
  float* pp0 = (float*)alloc(9L * 1024 * 1024 * 4);
  float* pp1 = (float*)alloc(9L * 1024 * 1024 * 4);
  const int stageC[5] = {128, 256, 512, 1024, 2048};
  float* stg[5];
  for (int i = 0; i < 5; ++i) stg[i] = (float*)alloc((long)NS[i] * stageC[i] * 4);

  // --- enc1_1: KPConv on raw 5-dim feats -> GN -> leaky --------------------
  // wf rows padded 75 -> 96 halves (matches padded weight), K%32==0.
  kpconv(F(IDX_PTS + 0), F(IDX_PTS + 0), I(IDX_NBR + 0), F(IDX_CONST + 0),
         F(IDX_FEATS), wfbf, NS[0], KSA[0], 5, 96, SIG1);
  gemm(wfbf, w_e11, t0, NS[0], 96, 64, 64, nullptr);
  gn(t0, pp0, F(IDX_E11_G), F(IDX_E11_B), nullptr, NS[0], 64, 1);
  float* cur = pp0;

  // --- encoder residual blocks --------------------------------------------
  for (int b = 0; b < 13; ++b) {
    const Blk& B = blk[b];
    const int Nin = NS[B.ss], Nq = NS[B.qs], K = KSA[B.ss], cm = B.cout / 4;
    const int* nb = B.strided ? I(IDX_SUB + B.ss) : I(IDX_NBR + B.qs);
    const float sigma = SIG1 * B.mult;
    const int wfK = MKP * cm;            // 480..7680, all %32==0

    // unary1: leaky(GN(cur @ u1_w))
    castf(cur, actbf, (long)Nin * B.cin);
    gemm(actbf, u1w[b], t0, Nin, B.cin, cm, cm, nullptr);
    gn(t0, t0, F(o_u1g[b]), F(o_u1b[b]), nullptr, Nin, cm, 1);

    // KPConv -> GN -> leaky
    kpconv(F(IDX_PTS + B.qs), F(IDX_PTS + B.ss), nb, F(IDX_CONST + 1 + b),
           t0, wfbf, Nq, K, cm, wfK, sigma);
    gemm(wfbf, kpw[b], t1, Nq, wfK, cm, cm, nullptr);
    gn(t1, t1, F(o_kpg[b]), F(o_kpb[b]), nullptr, Nq, cm, 1);

    // unary2: GN(x @ u2_w)  (no activation before residual add)
    castf(t1, actbf, (long)Nq * cm);
    gemm(actbf, u2w[b], t2, Nq, cm, B.cout, B.cout, nullptr);
    gn(t2, t2, F(o_u2g[b]), F(o_u2b[b]), nullptr, Nq, B.cout, 0);

    // shortcut
    const float* sc;
    if (B.strided) {                      // max-pool over neighbors (cin==cout)
      k_rowmax<<<Nq, 256, 0, stream>>>(cur, nb, t0, K, B.cin);
      sc = t0;
    } else if (B.has_sc) {                // projected shortcut
      castf(cur, actbf, (long)Nin * B.cin);
      gemm(actbf, scw[b], t0, Nq, B.cin, B.cout, B.cout, nullptr);
      gn(t0, t0, F(o_scg[b]), F(o_scb[b]), nullptr, Nq, B.cout, 0);
      sc = t0;
    } else {
      sc = cur;
    }

    float* dst = stageFinal[b] ? stg[B.qs] : ((cur == pp0) ? pp1 : pp0);
    k_add_leaky<<<gsz((long)Nq * B.cout), 256, 0, stream>>>(t2, sc, dst,
                                                            (long)Nq * B.cout);
    cur = dst;
  }

  // --- decoder: gather-upsample + concat + unary (leaky(GN(x@w))) ---------
  const float* lat = stg[4];
  int clat = 2048;
  for (int i = 3; i >= 0; --i) {
    const int Nq = NS[i], cskip = stageC[i];
    const int cin = clat + cskip, cout = cskip;
    const int db = IDX_DEC1 + 3 * i;   // b,g,w
    k_concat_gather<<<Nq, 256, 0, stream>>>(lat, I(IDX_UP + i), stg[i], actbf,
                                            clat, cskip);
    gemm(actbf, decw[i], t0, Nq, cin, cout, cout, nullptr);
    float* dst = (i & 1) ? pp0 : pp1;
    gn(t0, dst, F(db + 1), F(db + 0), nullptr, Nq, cout, 1);
    lat = dst; clat = cout;
  }

  // --- classifier head: relu(GN(lat@w1 + b1)) @ w2 + b2 -------------------
  castf(lat, actbf, (long)NS[0] * 128);
  gemm(actbf, w_cls1, t0, NS[0], 128, 64, 64, nullptr);
  gn(t0, t0, F(IDX_CLS_G), F(IDX_CLS_B), F(IDX_CLS_B1), NS[0], 64, 2);
  castf(t0, actbf, (long)NS[0] * 64);
  gemm(actbf, w_cls2, (float*)d_out, NS[0], 64, 16, 10, F(IDX_CLS_B2));
}
